// MultiheadAttention_721554506518
// MI455X (gfx1250) — matrix operand order repaired and checked by reading, not compiled, not run on hardware
//
#include <hip/hip_runtime.h>
#include <hip/hip_bf16.h>

typedef __bf16 bf16;
typedef __attribute__((ext_vector_type(16))) __bf16 v16bf;
typedef __attribute__((ext_vector_type(8)))  __bf16 bf16x8;
typedef __attribute__((ext_vector_type(8)))  float  v8f;

static constexpr int B_ = 4, S_ = 2048, D_ = 512, E_ = 512, H_ = 8, HD_ = 64;

// Async DMA: copy 16B/lane from global (sbase + voff bytes) into LDS at lds_off.
// Tracked by ASYNCcnt. GVS addressing mode: saddr = SGPR pair, vaddr = 32-bit.
__device__ __forceinline__ void async_ld_b128(uint32_t lds_off, uint32_t voff_bytes,
                                              const void* sbase)
{
    asm volatile("global_load_async_to_lds_b128 %0, %1, %2"
                 :: "v"(lds_off), "v"(voff_bytes), "s"(sbase)
                 : "memory");
}

// ---------------------------------------------------------------------------
// Pre-pass A: xb = bf16( x * (1-mask) ), flat [B*S, D]. 8 elems/thread.
// ---------------------------------------------------------------------------
__global__ __launch_bounds__(256) void cvt_x_kernel(
    const float* __restrict__ x, const int* __restrict__ mask,
    bf16* __restrict__ xb)
{
    const size_t idx = ((size_t)blockIdx.x * 256 + threadIdx.x) * 8;
    const size_t row = idx >> 9;                      // / D_ (=512)
    const float mf = 1.0f - (float)mask[row];
    const float* p = x + idx;
    bf16x8 v;
#pragma unroll
    for (int i = 0; i < 8; ++i) v[i] = (bf16)(p[i] * mf);
    *(bf16x8*)(xb + idx) = v;
}

// ---------------------------------------------------------------------------
// Pre-pass B: generic f32 -> bf16 convert, 8 elems/thread.
// ---------------------------------------------------------------------------
__global__ __launch_bounds__(256) void cvt_w_kernel(
    const float* __restrict__ src, bf16* __restrict__ dst)
{
    const size_t idx = ((size_t)blockIdx.x * 256 + threadIdx.x) * 8;
    const float* p = src + idx;
    bf16x8 v;
#pragma unroll
    for (int i = 0; i < 8; ++i) v[i] = (bf16)p[i];
    *(bf16x8*)(dst + idx) = v;
}

// ---------------------------------------------------------------------------
// Kernel 1: qkv = xb @ Wqkvb^T + bqkv. One wave = 32(M) x 64(N) tile:
// 2 A-fragments x 4 B-fragments -> 8 WMMAs per 32-wide K-chunk.
// A 64-wide N tile is exactly one {q|k|v} section of one head.
// Q,K stored [B,H,S,HD]; V stored transposed [B,H,HD,S].
// ---------------------------------------------------------------------------
__global__ __launch_bounds__(256) void qkv_gemm_kernel(
    const bf16* __restrict__ xb, const bf16* __restrict__ Wqkvb,
    const float* __restrict__ bqkv,
    bf16* __restrict__ Qb, bf16* __restrict__ Kb, bf16* __restrict__ Vt)
{
    const int lane = threadIdx.x & 31;
    const int wv   = threadIdx.x >> 5;
    const int wid  = blockIdx.x * 8 + wv;
    const int TN   = (3 * E_) / 64;                   // 24 N64-tiles
    const int tn   = wid % TN;
    const int tm   = wid / TN;
    const int m0   = tm * 32, n0 = tn * 64;
    const int half = lane >> 4, l16 = lane & 15;

    const bf16* arow0 = xb + (size_t)(m0 + l16) * D_;
    const bf16* arow1 = arow0 + (size_t)16 * D_;
    const bf16* wrow[4];
#pragma unroll
    for (int t = 0; t < 4; ++t)
        wrow[t] = Wqkvb + (size_t)(n0 + t * 16 + l16) * D_;

    const v8f vzero = {0.f,0.f,0.f,0.f,0.f,0.f,0.f,0.f};
    v8f acc[2][4];
#pragma unroll
    for (int mi = 0; mi < 2; ++mi)
#pragma unroll
        for (int t = 0; t < 4; ++t) acc[mi][t] = vzero;

    for (int k0 = 0; k0 < D_; k0 += 32) {
        const int ka = k0 + half * 16;                // A: K = ka..ka+15
        union { v16bf v; bf16x8 p[2]; } ua0, ua1;
        ua0.p[0] = *(const bf16x8*)(arow0 + ka);
        ua0.p[1] = *(const bf16x8*)(arow0 + ka + 8);
        ua1.p[0] = *(const bf16x8*)(arow1 + ka);
        ua1.p[1] = *(const bf16x8*)(arow1 + ka + 8);
        const int kb = k0 + half * 16;                // B: K = kb..kb+15
#pragma unroll
        for (int t = 0; t < 4; ++t) {
            const v16bf bfr = *(const v16bf*)(wrow[t] + kb);
            acc[0][t] = __builtin_amdgcn_wmma_f32_16x16x32_bf16(false, ua0.v, false, bfr,
                                                                (short)0, acc[0][t], false, false);
            acc[1][t] = __builtin_amdgcn_wmma_f32_16x16x32_bf16(false, ua1.v, false, bfr,
                                                                (short)0, acc[1][t], false, false);
        }
    }

    // Epilogue: per-wave scalars (tiles never straddle batch/section bounds)
    const int bb  = m0 >> 11;
    const int hh  = n0 / (3 * HD_);
    const int sec = (n0 >> 6) % 3;                    // 0=q, 1=k, 2=v

    if (sec == 2) {                                   // V -> transposed [B,H,HD,S]
        bf16* vb = Vt + ((size_t)bb * H_ + hh) * HD_ * S_;
#pragma unroll
        for (int mi = 0; mi < 2; ++mi) {
            const int s0 = (m0 & (S_ - 1)) + mi * 16;
#pragma unroll
            for (int t = 0; t < 4; ++t) {
                const int d = t * 16 + l16;
                const float bias = bqkv[n0 + d];
#pragma unroll
                for (int j = 0; j < 8; ++j)
                    vb[(size_t)d * S_ + s0 + j + 8 * half] = (bf16)(acc[mi][t][j] + bias);
            }
        }
    } else {                                          // Q or K -> [B,H,S,HD]
        bf16* dst = (sec ? Kb : Qb) + ((size_t)bb * H_ + hh) * S_ * HD_;
#pragma unroll
        for (int mi = 0; mi < 2; ++mi) {
            const int s0 = (m0 & (S_ - 1)) + mi * 16;
#pragma unroll
            for (int t = 0; t < 4; ++t) {
                const int d = t * 16 + l16;
                const float bias = bqkv[n0 + d];
#pragma unroll
                for (int j = 0; j < 8; ++j)
                    dst[(size_t)(s0 + j + 8 * half) * HD_ + d] = (bf16)(acc[mi][t][j] + bias);
            }
        }
    }
}

// ---------------------------------------------------------------------------
// Kernel 2: flash attention. Block = 8 waves sharing one (b,h), each wave one
// 16-query tile of a 128-query super-tile. K/V tiles (32 keys) are staged into
// double-buffered LDS by GLOBAL_LOAD_ASYNC_TO_LDS_B128 (ASYNCcnt), so the
// next chunk streams in while WMMAs consume the current one from LDS.
// ---------------------------------------------------------------------------
__global__ __launch_bounds__(256) void attn_kernel(
    const int* __restrict__ mask,
    const bf16* __restrict__ Qb, const bf16* __restrict__ Kb,
    const bf16* __restrict__ Vt, bf16* __restrict__ Valsb)
{
    __shared__ alignas(64) bf16 kbuf[2][32 * 64];     // K tile, [key][d], 2x4KB
    __shared__ alignas(64) bf16 vbuf[2][64 * 32];     // V tile, [d][key], 2x4KB
    __shared__ alignas(32) bf16 plds[8][16 * 32];     // per-wave P transpose, 8KB

    const int tid  = threadIdx.x;
    const int lane = tid & 31;
    const int wv   = tid >> 5;
    const int QB   = S_ / 128;                        // 16 query super-tiles
    const int qb   = blockIdx.x % QB;
    const int h    = (blockIdx.x / QB) % H_;
    const int b    = blockIdx.x / (QB * H_);
    const int q0   = qb * 128 + wv * 16;
    const int half = lane >> 4, l16 = lane & 15;
    const size_t headoff = ((size_t)b * H_ + h) * S_;
    const bf16* khead = Kb + headoff * HD_;           // [S,HD], 32-key tile contiguous
    const bf16* vhead = Vt + ((size_t)b * H_ + h) * HD_ * S_;  // [HD,S]
    const float NEGINF = -__builtin_inff();

    // per-thread async-copy coordinates (16B each)
    const uint32_t kgoff8 = (uint32_t)tid * 8;        // elem offset in 2048-elem K tile
    const int vd = tid >> 2, vk = (tid & 3) * 8;      // V: 4 threads per d-row

    // Q A-fragments: two 16x32 chunks covering HD=64
    v16bf aq[2];
    {
        const bf16* qrow = Qb + (headoff + (q0 + l16)) * HD_;
#pragma unroll
        for (int c = 0; c < 2; ++c) {
            const int d0 = c * 32 + half * 16;
            union { v16bf v; bf16x8 p[2]; } u;
            u.p[0] = *(const bf16x8*)(qrow + d0);
            u.p[1] = *(const bf16x8*)(qrow + d0 + 8);
            aq[c] = u.v;
        }
    }

    float mrow[8], lrow[8];
#pragma unroll
    for (int j = 0; j < 8; ++j) { mrow[j] = NEGINF; lrow[j] = 0.0f; }
    const v8f vzero = {0.f,0.f,0.f,0.f,0.f,0.f,0.f,0.f};
    v8f accd[4];
#pragma unroll
    for (int t = 0; t < 4; ++t) accd[t] = vzero;

    const float scale = 0.125f;                       // HD^-0.5
    const int* maskb = mask + b * S_;
    const int NIT = S_ / 32;                          // 64 chunks

    // prologue: stage chunk 0 into buffer 0
    async_ld_b128((uint32_t)(uintptr_t)&kbuf[0][kgoff8],
                  (uint32_t)((0 * HD_ + kgoff8) * sizeof(bf16)), khead);
    async_ld_b128((uint32_t)(uintptr_t)&vbuf[0][vd * 32 + vk],
                  (uint32_t)((vd * S_ + 0 + vk) * sizeof(bf16)), vhead);

    for (int it = 0; it < NIT; ++it) {
        const int k0  = it * 32;
        const int buf = it & 1;

        if (it + 1 < NIT) {                           // stage chunk it+1 -> other buffer
            const int kn = k0 + 32;
            async_ld_b128((uint32_t)(uintptr_t)&kbuf[buf ^ 1][kgoff8],
                          (uint32_t)((kn * HD_ + kgoff8) * sizeof(bf16)), khead);
            async_ld_b128((uint32_t)(uintptr_t)&vbuf[buf ^ 1][vd * 32 + vk],
                          (uint32_t)((vd * S_ + kn + vk) * sizeof(bf16)), vhead);
            asm volatile("s_wait_asynccnt 0x2" ::: "memory");   // chunk it landed
        } else {
            asm volatile("s_wait_asynccnt 0x0" ::: "memory");
        }
        __syncthreads();                              // all waves' portions visible

        // ---- logits for two 16-key tiles from LDS K tile
        v8f sl[2];
#pragma unroll
        for (int kt = 0; kt < 2; ++kt) {
            const int kl = kt * 16 + l16;             // B col = key (local)
            v8f acc = vzero;
#pragma unroll
            for (int c = 0; c < 2; ++c) {
                const v16bf bk = *(const v16bf*)&kbuf[buf][kl * 64 + c * 32 + half * 16];
                acc = __builtin_amdgcn_wmma_f32_16x16x32_bf16(false, aq[c], false, bk,
                                                              (short)0, acc, false, false);
            }
            const float mb = maskb[k0 + kt * 16 + l16] ? NEGINF : 0.0f;
#pragma unroll
            for (int j = 0; j < 8; ++j) acc[j] = acc[j] * scale + mb;
            sl[kt] = acc;
        }

        // ---- online softmax row stats
        float pl[8], ph[8], sc[8];
#pragma unroll
        for (int j = 0; j < 8; ++j) {
            float t = fmaxf(sl[0][j], sl[1][j]);
#pragma unroll
            for (int m = 1; m < 16; m <<= 1) t = fmaxf(t, __shfl_xor(t, m, 32));
            const float mnew = fmaxf(mrow[j], t);
            const float msub = (mnew == NEGINF) ? 0.0f : mnew;
            const float p0 = __expf(sl[0][j] - msub);
            const float p1 = __expf(sl[1][j] - msub);
            float rs = p0 + p1;
#pragma unroll
            for (int m = 1; m < 16; m <<= 1) rs += __shfl_xor(rs, m, 32);
            const float esc = (mrow[j] == NEGINF) ? 0.0f : __expf(mrow[j] - msub);
            lrow[j] = lrow[j] * esc + rs;
            mrow[j] = mnew;
            pl[j] = p0; ph[j] = p1; sc[j] = esc;
        }

        // ---- P: C-layout -> LDS -> A-layout (in-wave, in-order LDS)
        bf16* pw = &plds[wv][0];
#pragma unroll
        for (int j = 0; j < 8; ++j) {
            const int rr = j + 8 * half;
            pw[rr * 32 + l16]      = (bf16)pl[j];
            pw[rr * 32 + 16 + l16] = (bf16)ph[j];
        }
        asm volatile("s_wait_dscnt 0" ::: "memory");
        __builtin_amdgcn_wave_barrier();
        v16bf ap;
        {
            const bf16* pr = pw + l16 * 32 + half * 16;
            union { v16bf v; bf16x8 p[2]; } u;
            u.p[0] = *(const bf16x8*)(pr);
            u.p[1] = *(const bf16x8*)(pr + 8);
            ap = u.v;
        }

        // ---- rescale accumulators, then accd += P @ V from LDS V tile
#pragma unroll
        for (int t = 0; t < 4; ++t)
#pragma unroll
            for (int j = 0; j < 8; ++j) accd[t][j] *= sc[j];

#pragma unroll
        for (int t = 0; t < 4; ++t) {
            const v16bf bv = *(const v16bf*)&vbuf[buf][(t * 16 + l16) * 32 + half * 16];
            accd[t] = __builtin_amdgcn_wmma_f32_16x16x32_bf16(false, ap, false, bv,
                                                              (short)0, accd[t], false, false);
        }
        __syncthreads();                              // reads done before rebuffering
    }

    // ---- normalize and write vals [B,S,E] (e = h*HD + d) as bf16
#pragma unroll
    for (int j = 0; j < 8; ++j) {
        const float inv = 1.0f / lrow[j];
        const int srow = q0 + j + 8 * half;
        bf16* orow = Valsb + ((size_t)b * S_ + srow) * E_ + h * HD_;
#pragma unroll
        for (int t = 0; t < 4; ++t)
            orow[t * 16 + l16] = (bf16)(accd[t][j] * inv);
    }
}

// ---------------------------------------------------------------------------
// Kernel 3: out = vals @ Wob^T + bo (f32 out). One wave = 32(M) x 64(N).
// ---------------------------------------------------------------------------
__global__ __launch_bounds__(256) void out_gemm_kernel(
    const bf16* __restrict__ vals, const bf16* __restrict__ Wob,
    const float* __restrict__ bo, float* __restrict__ out)
{
    const int lane = threadIdx.x & 31;
    const int wv   = threadIdx.x >> 5;
    const int wid  = blockIdx.x * 8 + wv;
    const int TN   = E_ / 64;                         // 8 N64-tiles
    const int tn   = wid % TN;
    const int tm   = wid / TN;
    const int m0   = tm * 32, n0 = tn * 64;
    const int half = lane >> 4, l16 = lane & 15;

    const bf16* arow0 = vals + (size_t)(m0 + l16) * E_;
    const bf16* arow1 = arow0 + (size_t)16 * E_;
    const bf16* wrow[4];
#pragma unroll
    for (int t = 0; t < 4; ++t)
        wrow[t] = Wob + (size_t)(n0 + t * 16 + l16) * E_;

    const v8f vzero = {0.f,0.f,0.f,0.f,0.f,0.f,0.f,0.f};
    v8f acc[2][4];
#pragma unroll
    for (int mi = 0; mi < 2; ++mi)
#pragma unroll
        for (int t = 0; t < 4; ++t) acc[mi][t] = vzero;

    for (int k0 = 0; k0 < E_; k0 += 32) {
        const int ka = k0 + half * 16;
        union { v16bf v; bf16x8 p[2]; } ua0, ua1;
        ua0.p[0] = *(const bf16x8*)(arow0 + ka);
        ua0.p[1] = *(const bf16x8*)(arow0 + ka + 8);
        ua1.p[0] = *(const bf16x8*)(arow1 + ka);
        ua1.p[1] = *(const bf16x8*)(arow1 + ka + 8);
        const int kb = k0 + half * 16;
#pragma unroll
        for (int t = 0; t < 4; ++t) {
            const v16bf bfr = *(const v16bf*)(wrow[t] + kb);
            acc[0][t] = __builtin_amdgcn_wmma_f32_16x16x32_bf16(false, ua0.v, false, bfr,
                                                                (short)0, acc[0][t], false, false);
            acc[1][t] = __builtin_amdgcn_wmma_f32_16x16x32_bf16(false, ua1.v, false, bfr,
                                                                (short)0, acc[1][t], false, false);
        }
    }

#pragma unroll
    for (int mi = 0; mi < 2; ++mi)
#pragma unroll
        for (int t = 0; t < 4; ++t) {
            const int n = n0 + t * 16 + l16;
            const float bias = bo[n];
#pragma unroll
            for (int j = 0; j < 8; ++j)
                out[(size_t)(m0 + mi * 16 + j + 8 * half) * E_ + n] = acc[mi][t][j] + bias;
        }
}

// ---------------------------------------------------------------------------
extern "C" void kernel_launch(void* const* d_in, const int* in_sizes, int n_in,
                              void* d_out, int out_size, void* d_ws, size_t ws_size,
                              hipStream_t stream)
{
    const float* x    = (const float*)d_in[0];
    const int*   mask = (const int*)d_in[1];
    const float* Wqkv = (const float*)d_in[2];
    const float* bqkv = (const float*)d_in[3];
    const float* Wo   = (const float*)d_in[4];
    const float* bo   = (const float*)d_in[5];
    float* out = (float*)d_out;

    const size_t nX    = (size_t)B_ * S_ * D_;        // 4M
    const size_t nWqkv = (size_t)3 * E_ * D_;
    const size_t nWo   = (size_t)E_ * E_;
    const size_t nHead = (size_t)B_ * H_ * S_ * HD_;  // 4M

    bf16* xb    = (bf16*)d_ws;
    bf16* Wqkvb = xb + nX;
    bf16* Wob   = Wqkvb + nWqkv;
    bf16* Qb    = Wob + nWo;
    bf16* Kb    = Qb + nHead;
    bf16* Vt    = Kb + nHead;                         // transposed [B,H,HD,S]
    bf16* Valsb = Vt + nHead;                         // [B,S,E]
    // total ws: ~43 MB

    cvt_x_kernel<<<(int)(nX / 8 / 256), 256, 0, stream>>>(x, mask, xb);
    cvt_w_kernel<<<(int)(nWqkv / 8 / 256), 256, 0, stream>>>(Wqkv, Wqkvb);
    cvt_w_kernel<<<(int)(nWo / 8 / 256), 256, 0, stream>>>(Wo, Wob);

    {   // QKV GEMM: (8192/32) * (1536/64) = 6144 wave-tiles
        const int waves = (B_ * S_ / 32) * (3 * E_ / 64);
        qkv_gemm_kernel<<<waves / 8, 256, 0, stream>>>(xb, Wqkvb, bqkv, Qb, Kb, Vt);
    }
    {   // attention: one block per (b, h, 128-query super-tile) = 512 blocks
        const int blocks = B_ * H_ * (S_ / 128);
        attn_kernel<<<blocks, 256, 0, stream>>>(mask, Qb, Kb, Vt, Valsb);
    }
    {   // output GEMM: (8192/32) * (512/64) = 2048 wave-tiles
        const int waves = (B_ * S_ / 32) * (E_ / 64);
        out_gemm_kernel<<<waves / 8, 256, 0, stream>>>(Valsb, Wob, bo, out);
    }
}
